// MixtureOfExpertsLayer_7430293422492
// MI455X (gfx1250) — matrix operand order repaired and checked by reading, not compiled, not run on hardware
//
#include <hip/hip_runtime.h>
#include <hip/hip_bf16.h>
#include <math.h>

#define HIDDEN  1024
#define NEXP    8
#define EXPERT  2048
#define NTOK    8192

#define MT      32            // tokens per FFN block
#define FC      128           // ff-dim chunk
#define KHALF   512           // hidden staged in halves
#define XS      520           // X LDS row stride (halves), 1040B = 65*16 -> 16B aligned, bank-skewed
#define WS      136           // W tile row stride (halves), 272B = 17*16
#define HS      136           // H tile row stride (halves)
#define TILES_PER_E (NTOK / MT)   // 256

typedef __attribute__((ext_vector_type(16))) __bf16   v16bf;
typedef __attribute__((ext_vector_type(8)))  float    v8f;
typedef __attribute__((ext_vector_type(4)))  unsigned uint4v;   // trivial 16B vector (avoid HIP class uint4 in union)

union Frag32 { v16bf v; uint4v q[2]; };

// Round-to-nearest bf16 split: v ~= hi + lo with |v - hi - lo| ~ 2^-17 |v|
__device__ __forceinline__ void split_bf16(float v, unsigned short& hi, unsigned short& lo) {
    unsigned u  = __float_as_uint(v);
    unsigned rh = u + 0x7fffu + ((u >> 16) & 1u);
    hi = (unsigned short)(rh >> 16);
    float hf = __uint_as_float(((unsigned)hi) << 16);
    float l  = v - hf;
    unsigned u2 = __float_as_uint(l);
    unsigned rl = u2 + 0x7fffu + ((u2 >> 16) & 1u);
    lo = (unsigned short)(rl >> 16);
}

__device__ __forceinline__ v8f wmma_bf16(const Frag32& a, const Frag32& b, v8f c) {
    return __builtin_amdgcn_wmma_f32_16x16x32_bf16(false, a.v, false, b.v, (short)0, c, false, false);
}

// ---------------- kernel 1: zero output + counters ----------------
__global__ void moe_zero_kernel(float* __restrict__ out, int n, int* __restrict__ counters) {
    int i = blockIdx.x * blockDim.x + threadIdx.x;
    for (int j = i; j < n; j += gridDim.x * blockDim.x) out[j] = 0.f;
    if (i < 2 * NEXP) counters[i] = 0;   // counts[8] then cursors[8]
}

// ---------------- kernel 2: gating (wave per token) ----------------
__global__ void __launch_bounds__(256) moe_gate_kernel(
    const float* __restrict__ x, const float* __restrict__ gw, const float* __restrict__ gb,
    int* __restrict__ counts, int* __restrict__ gidx, float* __restrict__ gwt) {
    int gtid = blockIdx.x * blockDim.x + threadIdx.x;
    int tok  = gtid >> 5;
    int lane = gtid & 31;
    if (tok >= NTOK) return;
    const float* xr = x + (size_t)tok * HIDDEN;
    float acc[NEXP];
#pragma unroll
    for (int e = 0; e < NEXP; e++) acc[e] = 0.f;
    for (int h = lane; h < HIDDEN; h += 32) {
        float xv = xr[h];
        const float* g = gw + (size_t)h * NEXP;
#pragma unroll
        for (int e = 0; e < NEXP; e++) acc[e] = fmaf(xv, g[e], acc[e]);
    }
#pragma unroll
    for (int e = 0; e < NEXP; e++) {
#pragma unroll
        for (int m = 16; m >= 1; m >>= 1) acc[e] += __shfl_xor(acc[e], m, 32);
    }
    if (lane == 0) {
        float lg[NEXP];
#pragma unroll
        for (int e = 0; e < NEXP; e++) lg[e] = acc[e] + gb[e];
        int i0 = 0;
#pragma unroll
        for (int e = 1; e < NEXP; e++) if (lg[e] > lg[i0]) i0 = e;
        int i1 = (i0 == 0) ? 1 : 0;
#pragma unroll
        for (int e = 0; e < NEXP; e++) if (e != i0 && lg[e] > lg[i1]) i1 = e;
        // renormalized top-2 softmax weights: softmax denominator cancels
        float e1 = expf(lg[i1] - lg[i0]);
        float w0 = 1.f / (1.f + e1);
        float w1 = e1 / (1.f + e1);
        gidx[2 * tok]     = i0;  gidx[2 * tok + 1] = i1;
        gwt [2 * tok]     = w0;  gwt [2 * tok + 1] = w1;
        atomicAdd(&counts[i0], 1);
        atomicAdd(&counts[i1], 1);
    }
}

// ---------------- kernel 3: tiny exclusive scan over 8 experts ----------------
__global__ void moe_scan_kernel(const int* __restrict__ counts, int* __restrict__ offs,
                                int* __restrict__ cursors) {
    if (threadIdx.x == 0) {
        int s = 0;
        for (int e = 0; e < NEXP; e++) { offs[e] = s; s += counts[e]; }
    }
    if (threadIdx.x < NEXP) cursors[threadIdx.x] = 0;
}

// ---------------- kernel 4: scatter tokens into per-expert lists ----------------
__global__ void moe_scatter_kernel(const int* __restrict__ gidx, const float* __restrict__ gwt,
                                   const int* __restrict__ offs, int* __restrict__ cursors,
                                   int* __restrict__ rows, float* __restrict__ roww) {
    int t = blockIdx.x * blockDim.x + threadIdx.x;
    if (t >= NTOK) return;
#pragma unroll
    for (int k = 0; k < 2; k++) {
        int   e = gidx[2 * t + k];
        float w = gwt [2 * t + k];
        int   p = atomicAdd(&cursors[e], 1);
        rows[offs[e] + p] = t;
        roww[offs[e] + p] = w;
    }
}

// ---------------- kernel 5: fused expert FFN (bf16 hi/lo split WMMA) ----------------
__global__ void __launch_bounds__(256) moe_ffn_kernel(
    const float* __restrict__ x,  const float* __restrict__ w1, const float* __restrict__ b1,
    const float* __restrict__ w2, const float* __restrict__ b2,
    const int* __restrict__ rows, const float* __restrict__ roww,
    const int* __restrict__ counts, const int* __restrict__ offs,
    float* __restrict__ out) {
    const int e    = blockIdx.x / TILES_PER_E;
    const int tile = blockIdx.x % TILES_PER_E;
    const int cnt  = counts[e];
    if (tile * MT >= cnt) return;
    const int base = offs[e] + tile * MT;

    extern __shared__ char smem[];
    unsigned short* Xh = (unsigned short*)smem;        // [MT][XS]
    unsigned short* Xl = Xh + MT * XS;
    unsigned short* Wh = Xl + MT * XS;                 // [FC][WS]  (shared by w1t / w2t phases)
    unsigned short* Wl = Wh + FC * WS;
    unsigned short* Hh = Wl + FC * WS;                 // [MT][HS]
    unsigned short* Hl = Hh + MT * HS;
    int*   rows_s = (int*)(Hl + MT * HS);
    float* wts_s  = (float*)(rows_s + MT);

    const int tid  = threadIdx.x;
    const int lane = tid & 31;
    const int wv   = tid >> 5;         // 8 waves
    const int ln16 = lane & 15;
    const int kg   = lane >> 4;        // K-group within fragment layouts
    const int nloc = wv * 16 + ln16;   // this wave's N column (0..127 within a 128-wide panel)

    if (tid < MT) {
        int ok = (tile * MT + tid) < cnt;
        rows_s[tid] = ok ? rows[base + tid] : -1;
        wts_s[tid]  = ok ? roww[base + tid] : 0.f;
    }
    __syncthreads();

    const float* W1e = w1 + (size_t)e * HIDDEN * EXPERT;   // [h][f]
    const float* W2e = w2 + (size_t)e * EXPERT * HIDDEN;   // [f][h]

    v8f Cy[2][8];
#pragma unroll
    for (int m = 0; m < 2; m++)
#pragma unroll
        for (int nb = 0; nb < 8; nb++)
            Cy[m][nb] = (v8f){0.f,0.f,0.f,0.f,0.f,0.f,0.f,0.f};

    for (int fc = 0; fc < EXPERT / FC; fc++) {     // 16 ff chunks
        const int f0 = fc * FC;
        v8f Ch[2];
        Ch[0] = (v8f){0.f,0.f,0.f,0.f,0.f,0.f,0.f,0.f};
        Ch[1] = (v8f){0.f,0.f,0.f,0.f,0.f,0.f,0.f,0.f};

        // ---- GEMM1: Hc(32x128) = X(32x1024) * w1[:, f0:f0+128] ----
        for (int khalf = 0; khalf < 2; khalf++) {
            // stage X half (gathered rows), split into bf16 hi/lo
            for (int i = tid; i < MT * KHALF; i += 256) {
                int r = i >> 9, c = i & (KHALF - 1);
                int tok = rows_s[r];
                float v = (tok >= 0) ? x[(size_t)tok * HIDDEN + khalf * KHALF + c] : 0.f;
                unsigned short h16, l16; split_bf16(v, h16, l16);
                Xh[r * XS + c] = h16;  Xl[r * XS + c] = l16;
            }
            __syncthreads();
            for (int kb = 0; kb < KHALF / FC; kb++) {   // 4 x (K=128) stages
                const int h0 = khalf * KHALF + kb * FC;
                // stage w1 tile transposed: Wh[f][h] = w1[h0+h][f0+f]  (coalesced read over f)
                for (int i = tid; i < FC * FC; i += 256) {
                    int hl = i >> 7, fl = i & 127;
                    float v = W1e[(size_t)(h0 + hl) * EXPERT + f0 + fl];
                    unsigned short h16, l16; split_bf16(v, h16, l16);
                    Wh[fl * WS + hl] = h16;  Wl[fl * WS + hl] = l16;
                }
                __syncthreads();
#pragma unroll
                for (int kk = 0; kk < 4; kk++) {        // K=32 WMMA steps
                    Frag32 Bh, Bl;
                    const unsigned short* bp  = Wh + nloc * WS + kk * 32 + kg * 8;
                    const unsigned short* bp2 = Wl + nloc * WS + kk * 32 + kg * 8;
                    Bh.q[0] = *(const uint4v*)(bp);    Bh.q[1] = *(const uint4v*)(bp + 16);
                    Bl.q[0] = *(const uint4v*)(bp2);   Bl.q[1] = *(const uint4v*)(bp2 + 16);
#pragma unroll
                    for (int m = 0; m < 2; m++) {
                        const int ao = (m * 16 + ln16) * XS + kb * FC + kk * 32 + kg * 8;
                        Frag32 Ah, Al;
                        Ah.q[0] = *(const uint4v*)(Xh + ao);
                        Ah.q[1] = *(const uint4v*)(Xh + ao + 16);
                        Al.q[0] = *(const uint4v*)(Xl + ao);
                        Al.q[1] = *(const uint4v*)(Xl + ao + 16);
                        Ch[m] = wmma_bf16(Ah, Bh, Ch[m]);   // hi*hi
                        Ch[m] = wmma_bf16(Ah, Bl, Ch[m]);   // hi*lo
                        Ch[m] = wmma_bf16(Al, Bh, Ch[m]);   // lo*hi
                    }
                }
                __syncthreads();
            }
        }

        // ---- bias + ReLU, split H to bf16 hi/lo A-tiles in LDS ----
        {
            float b1v = b1[e * EXPERT + f0 + nloc];
#pragma unroll
            for (int m = 0; m < 2; m++) {
#pragma unroll
                for (int r = 0; r < 8; r++) {
                    float hv = Ch[m][r] + b1v;
                    hv = hv > 0.f ? hv : 0.f;
                    unsigned short h16, l16; split_bf16(hv, h16, l16);
                    int row = m * 16 + r + 8 * kg;      // C/D layout: M = r + 8*(lane>=16)
                    Hh[row * HS + nloc] = h16;
                    Hl[row * HS + nloc] = l16;
                }
            }
        }
        __syncthreads();

        // ---- GEMM2: Y(32x1024) += Hc(32x128) * w2[f0:f0+128, :] ----
#pragma unroll
        for (int nb = 0; nb < 8; nb++) {
            // stage w2 tile transposed: Wh[h][f] = w2[f0+f][nb*128+h]  (coalesced read over h)
            for (int i = tid; i < FC * FC; i += 256) {
                int fl = i >> 7, hlc = i & 127;
                float v = W2e[(size_t)(f0 + fl) * HIDDEN + nb * 128 + hlc];
                unsigned short h16, l16; split_bf16(v, h16, l16);
                Wh[hlc * WS + fl] = h16;  Wl[hlc * WS + fl] = l16;
            }
            __syncthreads();
#pragma unroll
            for (int kk = 0; kk < 4; kk++) {
                Frag32 Bh, Bl;
                const unsigned short* bp  = Wh + nloc * WS + kk * 32 + kg * 8;
                const unsigned short* bp2 = Wl + nloc * WS + kk * 32 + kg * 8;
                Bh.q[0] = *(const uint4v*)(bp);    Bh.q[1] = *(const uint4v*)(bp + 16);
                Bl.q[0] = *(const uint4v*)(bp2);   Bl.q[1] = *(const uint4v*)(bp2 + 16);
#pragma unroll
                for (int m = 0; m < 2; m++) {
                    const int ao = (m * 16 + ln16) * HS + kk * 32 + kg * 8;
                    Frag32 Ah, Al;
                    Ah.q[0] = *(const uint4v*)(Hh + ao);
                    Ah.q[1] = *(const uint4v*)(Hh + ao + 16);
                    Al.q[0] = *(const uint4v*)(Hl + ao);
                    Al.q[1] = *(const uint4v*)(Hl + ao + 16);
                    Cy[m][nb] = wmma_bf16(Ah, Bh, Cy[m][nb]);
                    Cy[m][nb] = wmma_bf16(Ah, Bl, Cy[m][nb]);
                    Cy[m][nb] = wmma_bf16(Al, Bh, Cy[m][nb]);
                }
            }
            __syncthreads();
        }
    }

    // ---- weighted scatter-add: out[tok] += w_t * (Y + b2) ----
#pragma unroll
    for (int m = 0; m < 2; m++) {
#pragma unroll
        for (int r = 0; r < 8; r++) {
            int slot = m * 16 + r + 8 * kg;
            int tok  = rows_s[slot];
            float wt = wts_s[slot];
            if (tok >= 0) {
#pragma unroll
                for (int nb = 0; nb < 8; nb++) {
                    int h = nb * 128 + nloc;
                    float val = wt * (Cy[m][nb][r] + b2[e * HIDDEN + h]);
                    atomicAdd(&out[(size_t)tok * HIDDEN + h], val);
                }
            }
        }
    }
}

// ---------------- host launcher ----------------
extern "C" void kernel_launch(void* const* d_in, const int* in_sizes, int n_in,
                              void* d_out, int out_size, void* d_ws, size_t ws_size,
                              hipStream_t stream) {
    (void)in_sizes; (void)n_in; (void)out_size; (void)ws_size;
    const float* x   = (const float*)d_in[0];
    const float* gw  = (const float*)d_in[1];
    const float* gb  = (const float*)d_in[2];
    const float* w1  = (const float*)d_in[3];
    const float* b1  = (const float*)d_in[4];
    const float* w2  = (const float*)d_in[5];
    const float* b2  = (const float*)d_in[6];
    float* out = (float*)d_out;

    // workspace layout (all 4-byte aligned; ~256 KB total)
    int*   counts  = (int*)d_ws;            // 8
    int*   cursors = counts + 8;            // 8
    int*   offs    = counts + 16;           // 8 (+8 pad)
    int*   gidx    = counts + 32;           // 2*NTOK
    float* gwt     = (float*)(gidx + 2 * NTOK);
    int*   rows    = (int*)(gwt + 2 * NTOK);   // 2*NTOK
    float* roww    = (float*)(rows + 2 * NTOK);

    constexpr int SMEM_BYTES = (2 * (MT * XS) + 2 * (FC * WS) + 2 * (MT * HS)) * 2 + MT * 8; // 153856
    (void)hipFuncSetAttribute((const void*)moe_ffn_kernel,
                              hipFuncAttributeMaxDynamicSharedMemorySize, SMEM_BYTES);

    moe_zero_kernel<<<8192, 256, 0, stream>>>(out, NTOK * HIDDEN, counts);
    moe_gate_kernel<<<(NTOK * 32) / 256, 256, 0, stream>>>(x, gw, gb, counts, gidx, gwt);
    moe_scan_kernel<<<1, 32, 0, stream>>>(counts, offs, cursors);
    moe_scatter_kernel<<<(NTOK + 255) / 256, 256, 0, stream>>>(gidx, gwt, offs, cursors, rows, roww);
    moe_ffn_kernel<<<NEXP * TILES_PER_E, 256, SMEM_BYTES, stream>>>(
        x, w1, b1, w2, b2, rows, roww, counts, offs, out);
}
